// ConvGeodesic_42442866819429
// MI455X (gfx1250) — compile-verified, matrix operand order checked by reading, not run on hardware
//
#include <hip/hip_runtime.h>
#include <hip/hip_bf16.h>
#include <stdint.h>

// ---------------- problem constants (from reference) ----------------
#define B_    4
#define V_    50000
#define F_    64
#define G_    2048
#define R_    5
#define ANG_  8
#define NROT_ 8
#define O_    64
#define M_    (B_*G_)          // 8192 rows  (b,g)
#define KTOT_ (R_*ANG_*F_)     // 2560 reduction dim (r,a,f)
#define N_    (NROT_*O_)       // 512 cols  (n,o)
#define KT_   (KTOT_/32)       // 80 K-tiles of 32

typedef __attribute__((ext_vector_type(16))) _Float16 v16h;
typedef __attribute__((ext_vector_type(8)))  float    v8f;

// pack two floats to two f16 in a u32
__device__ __forceinline__ unsigned pack2h(float x, float y) {
    unsigned short hx = __builtin_bit_cast(unsigned short, (_Float16)x);
    unsigned short hy = __builtin_bit_cast(unsigned short, (_Float16)y);
    return (unsigned)hx | ((unsigned)hy << 16);
}

// WMMA 16-bit operand lane layout (16x32 tile, wave32):
//   lane L, element e (0..15):  m/n = L%16,
//   k_local = (e%8) + (e/8)*16 + (L/16)*8
// Fragment stored contiguously per lane: 16 f16 = 32 bytes -> one v16h load.
// Swizzled index: ((tile16 * KT + kTile)*32 + lane)*16 + e

// ---------------- phase 1: gather + barycentric interp -> Xsw (f16, swizzled)
__global__ __launch_bounds__(256)
void ConvGeo_gather_interp(const float* __restrict__ signal,
                           const int*   __restrict__ bidx,
                           const float* __restrict__ bw,
                           unsigned short* __restrict__ Xsw)
{
    int tid = blockIdx.x * blockDim.x + threadIdx.x;   // < 327680*32
    int lane = tid & 31;          // lane handles f pair (2*lane, 2*lane+1)
    int t    = tid >> 5;          // tuple (b,g,r,a)
    if (t >= B_*G_*R_*ANG_) return;
    int a = t & 7;
    int r = (t >> 3) % R_;
    int g = (t / (R_*ANG_)) & (G_-1);
    int b = t / (R_*ANG_*G_);

    int base3 = t * 3;
    int i0 = bidx[base3+0], i1 = bidx[base3+1], i2 = bidx[base3+2];
    float w0 = bw[base3+0], w1 = bw[base3+1], w2 = bw[base3+2];

    int f0 = lane * 2;
    const float2* s2 = (const float2*)signal;
    float2 s0  = s2[((b*V_ + i0)*F_ + f0) >> 1];
    float2 s1  = s2[((b*V_ + i1)*F_ + f0) >> 1];
    float2 s2v = s2[((b*V_ + i2)*F_ + f0) >> 1];

    float v0 = w0*s0.x + w1*s1.x + w2*s2v.x;
    float v1 = w0*s0.y + w1*s1.y + w2*s2v.y;

    int row = b*G_ + g;
    int kk  = (r*ANG_ + a)*F_ + f0;            // even
    int rowTile = row >> 4, m = row & 15;
    int kTile = kk >> 5, kl = kk & 31;         // kl even -> e even
    int laneSw = m + (((kl >> 3) & 1) << 4);
    int e      = (kl & 7) + ((kl >> 4) << 3);
    int off = ((rowTile*KT_ + kTile)*32 + laneSw)*16 + e;   // u16 units, even
    ((unsigned*)Xsw)[off >> 1] = pack2h(v0, v1);
}

// ---------------- phase 2: fold rotations + K-sum into swizzled f16 weights
// Wbig[(r,a'',f),(n,o)] = sum_k kernel[r,(a''+n)%A, k, o, f]
__global__ __launch_bounds__(256)
void ConvGeo_prep_weights(const float* __restrict__ kern,
                          unsigned short* __restrict__ Wsw)
{
    int id = blockIdx.x * blockDim.x + threadIdx.x;   // < 32*KT_*512
    if (id >= (N_/16)*KT_*512) return;
    int e     = id & 15;
    int lane  = (id >> 4) & 31;
    int kTile = (id >> 9) % KT_;
    int nTile = id / (512*KT_);

    int n  = lane & 15;
    int kl = (e & 7) + ((e >> 3) << 4) + ((lane >> 4) << 3);
    int col  = nTile*16 + n;
    int nrot = col >> 6;
    int o    = col & 63;
    int kk = kTile*32 + kl;
    int r  = kk >> 9;            // /(A*F)
    int a2 = (kk >> 6) & 7;
    int f  = kk & 63;
    int arot = (a2 + nrot) & 7;
    int kb = (((r*ANG_ + arot)*2 + 0)*O_ + o)*F_ + f;   // kernel (R,A,K,O,F)
    float v = kern[kb] + kern[kb + O_*F_];              // sum over K=2
    Wsw[id] = __builtin_bit_cast(unsigned short, (_Float16)v);
}

// ---------------- phase 3: GEMM 8192x512x2560 via v_wmma_f32_16x16x32_f16
// block = 4 waves; wave w owns rows [blockIdx.x*64 + w*16, +16) x 64 cols.
// Software-pipelined: fragments for step kt+1 are loaded while WMMAs for
// step kt execute, so the pre-WMMA s_wait covers loads issued a full
// iteration earlier.
#define WMMA16(d, a, b) \
    d = __builtin_amdgcn_wmma_f32_16x16x32_f16(false, a, false, b, \
                                               (short)0, d, false, false)

__global__ __launch_bounds__(128)
void ConvGeo_gemm_wmma(const unsigned short* __restrict__ Xsw_,
                       const unsigned short* __restrict__ Wsw_,
                       const float* __restrict__ bias,
                       float* __restrict__ out)
{
    const v16h* Xsw = (const v16h*)Xsw_;
    const v16h* Wsw = (const v16h*)Wsw_;
    int wave = threadIdx.x >> 5;
    int lane = threadIdx.x & 31;
    int rowTile = blockIdx.x*4 + wave;     // 16-row tile index, 0..511
    int nTile0  = blockIdx.y*4;            // 16-col tile base, 0..31

    const v16h* aP = Xsw + (size_t)rowTile*KT_*32 + lane;
    const v16h* b0 = Wsw + (size_t)(nTile0+0)*KT_*32 + lane;
    const v16h* b1 = Wsw + (size_t)(nTile0+1)*KT_*32 + lane;
    const v16h* b2 = Wsw + (size_t)(nTile0+2)*KT_*32 + lane;
    const v16h* b3 = Wsw + (size_t)(nTile0+3)*KT_*32 + lane;

    v8f acc0 = {}, acc1 = {}, acc2 = {}, acc3 = {};

    // prologue: stage kt=0 fragments
    v16h aF  = aP[0];
    v16h bF0 = b0[0], bF1 = b1[0], bF2 = b2[0], bF3 = b3[0];

    for (int kt = 0; kt < KT_ - 1; ++kt) {
        size_t nx = (size_t)(kt + 1) * 32;
        // issue next-step loads first ...
        v16h aN  = aP[nx];
        v16h bN0 = b0[nx];
        v16h bN1 = b1[nx];
        v16h bN2 = b2[nx];
        v16h bN3 = b3[nx];
        if (kt + 2 < KT_) __builtin_prefetch(aP + (size_t)(kt + 2)*32, 0, 1);
        // ... then consume current fragments (loaded last iteration)
        WMMA16(acc0, aF, bF0);
        WMMA16(acc1, aF, bF1);
        WMMA16(acc2, aF, bF2);
        WMMA16(acc3, aF, bF3);
        aF = aN; bF0 = bN0; bF1 = bN1; bF2 = bN2; bF3 = bN3;
    }
    // epilogue step
    WMMA16(acc0, aF, bF0);
    WMMA16(acc1, aF, bF1);
    WMMA16(acc2, aF, bF2);
    WMMA16(acc3, aF, bF3);

    // writeout: C/D layout lane L, vgpr j -> row = 16*rowTile + j + 8*(L/16),
    // col = 16*tile + L%16.  bias term = (R*A) * sum_k bias[k,o] = 40*(b0+b1).
    int ncol = lane & 15;
    int rsub = (lane >> 4) << 3;
#pragma unroll
    for (int t = 0; t < 4; ++t) {
        v8f acc = (t==0) ? acc0 : (t==1) ? acc1 : (t==2) ? acc2 : acc3;
        int col  = (nTile0 + t)*16 + ncol;
        int nrot = col >> 6;
        int o    = col & 63;
        float bs = 40.0f * (bias[o] + bias[O_ + o]);
#pragma unroll
        for (int j = 0; j < 8; ++j) {
            int row = rowTile*16 + j + rsub;
            int b = row >> 11;          // /G
            int g = row & (G_-1);
            size_t oi = ((size_t)(b*NROT_ + nrot)*G_ + g)*O_ + o;
            out[oi] = acc[j] + bs;
        }
    }
}

// ---------------- launch ----------------
extern "C" void kernel_launch(void* const* d_in, const int* in_sizes, int n_in,
                              void* d_out, int out_size, void* d_ws, size_t ws_size,
                              hipStream_t stream) {
    const float* signal = (const float*)d_in[0];
    const int*   bidx   = (const int*)  d_in[1];
    const float* bw     = (const float*)d_in[2];
    const float* kern   = (const float*)d_in[3];
    const float* bias   = (const float*)d_in[4];
    float* out = (float*)d_out;

    unsigned short* Xsw = (unsigned short*)d_ws;           // 8192*2560 f16 = 41.9 MB
    unsigned short* Wsw = Xsw + (size_t)M_ * KTOT_;        // 512*2560  f16 =  2.6 MB

    {   // phase 1: 327680 tuples * 32 lanes
        int total  = B_*G_*R_*ANG_*32;
        ConvGeo_gather_interp<<<(total + 255)/256, 256, 0, stream>>>(
            signal, bidx, bw, Xsw);
    }
    {   // phase 2: 32 nTiles * 80 kTiles * 512 slots
        int total = (N_/16)*KT_*512;
        ConvGeo_prep_weights<<<(total + 255)/256, 256, 0, stream>>>(kern, Wsw);
    }
    {   // phase 3: 64x64 output tile per block
        dim3 grid(M_/64, N_/64);   // (128, 8)
        ConvGeo_gemm_wmma<<<grid, dim3(128), 0, stream>>>(Xsw, Wsw, bias, out);
    }
}
